// VectorQuantizeEMA_12086037971138
// MI455X (gfx1250) — compile-verified, matrix operand order checked by reading, not run on hardware
//
#include <hip/hip_runtime.h>
#include <hip/hip_bf16.h>
#include <math.h>

// ---------------- Types for CDNA5 WMMA ----------------
typedef __attribute__((ext_vector_type(16))) __bf16 v16bf;
typedef __attribute__((ext_vector_type(8)))  __bf16 v8bf;
typedef __attribute__((ext_vector_type(8)))  float  v8f;

#define NT 32768      // tokens
#define DIMD 512      // latent dim
#define NC 4096       // codes
#define DECAY 0.995f
#define EPSF 0.0001f
#define THRESH 0.99f

// Output layout (flat f32 concat in reference return order)
#define OUT_Z      ((size_t)0)
#define OUT_DIFF   ((size_t)16777216)
#define OUT_CODES  ((size_t)16777217)
#define OUT_EMB    ((size_t)16809985)
#define OUT_SIZE   ((size_t)18907137)
#define OUT_SUM    ((size_t)18911233)
#define OUT_AVG    ((size_t)21008385)
#define OUT_USAGE  ((size_t)21008386)
#define OUT_ENT    ((size_t)21008387)

__device__ inline v8f wmma_bf16(v16bf a, v16bf b, v8f c) {
    return __builtin_amdgcn_wmma_f32_16x16x32_bf16(
        /*neg_a=*/false, a, /*neg_b=*/false, b,
        /*c_mod=*/(short)0, c, /*reuse_a=*/false, /*reuse_b=*/false);
}

// Per-lane fragment gather for 16-bit A(16x32) / B(32x16) (ISA 7.12.2):
// lane l holds row l%16; lanes 0-15: K={k..k+7, k+16..k+23}, lanes 16-31: +8.
__device__ inline v16bf load_frag(const __bf16* __restrict__ rowBase, int k, int half) {
    const v8bf* p0 = (const v8bf*)(rowBase + k + half);
    const v8bf* p1 = (const v8bf*)(rowBase + k + half + 16);
    v8bf x0 = *p0, x1 = *p1;
    v16bf r;
#pragma unroll
    for (int i = 0; i < 8; ++i) { r[i] = x0[i]; r[i + 8] = x1[i]; }
    return r;
}

// ---------------- Prep kernels ----------------
__global__ void k_zero(float* __restrict__ p, int n) {
    int i = blockIdx.x * blockDim.x + threadIdx.x;
    if (i < n) p[i] = 0.0f;
}

// f32 -> (hi, lo) bf16 split for near-f32 WMMA accuracy
__global__ void k_split(const float* __restrict__ src, __bf16* __restrict__ hi,
                        __bf16* __restrict__ lo, int n) {
    int i = blockIdx.x * blockDim.x + threadIdx.x;
    if (i < n) {
        float f = src[i];
        __bf16 h = (__bf16)f;
        hi[i] = h;
        lo[i] = (__bf16)(f - (float)h);
    }
}

__global__ void k_enorm(const float* __restrict__ e, float* __restrict__ enorm) {
    __shared__ float red[128];
    int c = blockIdx.x;
    float s = 0.0f;
    for (int d = threadIdx.x; d < DIMD; d += 128) {
        float v = e[(size_t)c * DIMD + d];
        s += v * v;
    }
    red[threadIdx.x] = s;
    __syncthreads();
    for (int off = 64; off > 0; off >>= 1) {
        if (threadIdx.x < off) red[threadIdx.x] += red[threadIdx.x + off];
        __syncthreads();
    }
    if (threadIdx.x == 0) enorm[c] = red[0];
}

// ---------------- Fused GEMM + argmin (the WMMA kernel) ----------------
// 256 threads = 8 waves; wave w owns tokens [blk*128 + w*16, +16).
// Loops over all 4096 codes in pairs of 16-column tiles; K=512 in steps of 32.
// score = ||e||^2 - 2*(x.e), x.e ~= xh*eh + xh*el + xl*eh (3 WMMAs).
__global__ void __launch_bounds__(256)
k_gemm_argmin(const __bf16* __restrict__ xh, const __bf16* __restrict__ xl,
              const __bf16* __restrict__ eh, const __bf16* __restrict__ el,
              const float* __restrict__ enorm, int* __restrict__ out_idx) {
    __shared__ float s_val[8][32][8];
    __shared__ int   s_idx[8][32][8];

    const int lane    = threadIdx.x & 31;
    const int wave    = threadIdx.x >> 5;
    const int laneRow = lane & 15;
    const int half    = (lane >> 4) * 8;
    const int tokBase = blockIdx.x * 128 + wave * 16;

    const __bf16* aH = xh + (size_t)(tokBase + laneRow) * DIMD;
    const __bf16* aL = xl + (size_t)(tokBase + laneRow) * DIMD;

    float bestV[8];
    int   bestI[8];
#pragma unroll
    for (int r = 0; r < 8; ++r) { bestV[r] = 3.4e38f; bestI[r] = 0; }

    for (int p = 0; p < NC / 32; ++p) {
        const int c0 = p * 32;
        const int c1 = c0 + 16;
        const __bf16* bH0 = eh + (size_t)(c0 + laneRow) * DIMD;
        const __bf16* bL0 = el + (size_t)(c0 + laneRow) * DIMD;
        const __bf16* bH1 = eh + (size_t)(c1 + laneRow) * DIMD;
        const __bf16* bL1 = el + (size_t)(c1 + laneRow) * DIMD;

        v8f acc0 = {};
        v8f acc1 = {};
#pragma unroll 4
        for (int ks = 0; ks < DIMD; ks += 32) {
            v16bf a_h  = load_frag(aH,  ks, half);
            v16bf a_l  = load_frag(aL,  ks, half);
            v16bf b_h0 = load_frag(bH0, ks, half);
            v16bf b_l0 = load_frag(bL0, ks, half);
            v16bf b_h1 = load_frag(bH1, ks, half);
            v16bf b_l1 = load_frag(bL1, ks, half);
            acc0 = wmma_bf16(a_h, b_h0, acc0);
            acc0 = wmma_bf16(a_h, b_l0, acc0);
            acc0 = wmma_bf16(a_l, b_h0, acc0);
            acc1 = wmma_bf16(a_h, b_h1, acc1);
            acc1 = wmma_bf16(a_h, b_l1, acc1);
            acc1 = wmma_bf16(a_l, b_h1, acc1);
        }
        float en0 = enorm[c0 + laneRow];
        float en1 = enorm[c1 + laneRow];
#pragma unroll
        for (int r = 0; r < 8; ++r) {
            float s0 = en0 - 2.0f * acc0[r];
            if (s0 < bestV[r]) { bestV[r] = s0; bestI[r] = c0 + laneRow; }
            float s1 = en1 - 2.0f * acc1[r];
            if (s1 < bestV[r]) { bestV[r] = s1; bestI[r] = c1 + laneRow; }
        }
    }

#pragma unroll
    for (int r = 0; r < 8; ++r) {
        s_val[wave][lane][r] = bestV[r];
        s_idx[wave][lane][r] = bestI[r];
    }
    __syncthreads();

    // C-layout (ISA 7.12.2): VGPR r, lanes 0-15 -> M=r, lanes 16-31 -> M=r+8.
    if (threadIdx.x < 128) {
        int w = threadIdx.x >> 4;
        int m = threadIdx.x & 15;
        int laneBase = (m >= 8) ? 16 : 0;
        int r = m & 7;
        float bv = 3.4e38f;
        int   bi = 0;
        for (int n = 0; n < 16; ++n) {
            float v = s_val[w][laneBase + n][r];
            int   i = s_idx[w][laneBase + n][r];
            if (v < bv) { bv = v; bi = i; }
        }
        out_idx[blockIdx.x * 128 + w * 16 + m] = bi;
    }
}

// ---------------- Gather/scatter: z, codes, diff, csize, csum ----------------
__global__ void k_scatter(const float* __restrict__ x, const float* __restrict__ emb,
                          const int* __restrict__ idx, float* __restrict__ z,
                          float* __restrict__ codes, float* __restrict__ csize,
                          float* __restrict__ csum, float* __restrict__ scal) {
    __shared__ int s_code;
    __shared__ float red[128];
    int t = blockIdx.x;
    if (threadIdx.x == 0) {
        int c = idx[t];
        s_code = c;
        codes[t] = (float)c;
        atomicAdd(&csize[c], 1.0f);
    }
    __syncthreads();
    int c = s_code;
    float local = 0.0f;
#pragma unroll
    for (int j = 0; j < 4; ++j) {
        int d = threadIdx.x + j * 128;
        float q  = emb[(size_t)c * DIMD + d];
        float xv = x[(size_t)t * DIMD + d];
        z[(size_t)t * DIMD + d] = xv + (q - xv);   // matches reference rounding
        float dd = q - xv;
        local += dd * dd;
        atomicAdd(&csum[(size_t)c * DIMD + d], xv);
    }
    red[threadIdx.x] = local;
    __syncthreads();
    for (int off = 64; off > 0; off >>= 1) {
        if (threadIdx.x < off) red[threadIdx.x] += red[threadIdx.x + off];
        __syncthreads();
    }
    if (threadIdx.x == 0) atomicAdd(&scal[0], red[0]);  // diff accumulator
}

// ---------------- Per-code stats: new_size, n, usage, entropy ----------------
__global__ void k_stats(const float* __restrict__ cluster_size, const float* __restrict__ csize,
                        float* __restrict__ out_newsize, float* __restrict__ scal) {
    __shared__ float rn[256], ru[256], re[256];
    int c = blockIdx.x * 256 + threadIdx.x;
    float ns = DECAY * cluster_size[c] + (1.0f - DECAY) * csize[c];
    out_newsize[c] = ns;
    float used = (ns >= THRESH) ? 1.0f : 0.0f;
    float pr = csize[c] * (1.0f / (float)NT);   // csize sums to NT exactly
    float ent = -pr * logf(pr + 1e-5f);
    rn[threadIdx.x] = ns;
    ru[threadIdx.x] = used;
    re[threadIdx.x] = ent;
    __syncthreads();
    for (int off = 128; off > 0; off >>= 1) {
        if (threadIdx.x < off) {
            rn[threadIdx.x] += rn[threadIdx.x + off];
            ru[threadIdx.x] += ru[threadIdx.x + off];
            re[threadIdx.x] += re[threadIdx.x + off];
        }
        __syncthreads();
    }
    if (threadIdx.x == 0) {
        atomicAdd(&scal[1], rn[0]);  // n
        atomicAdd(&scal[2], ru[0]);  // usage
        atomicAdd(&scal[3], re[0]);  // entropy
    }
}

// ---------------- EMA update: new_sum, new_embedding ----------------
__global__ void k_update(const float* __restrict__ cluster_sum, const float* __restrict__ csum,
                         const float* __restrict__ newsize, const float* __restrict__ x,
                         const float* __restrict__ scal,
                         float* __restrict__ out_newsum, float* __restrict__ out_newemb) {
    int c = blockIdx.x;
    float n  = scal[1];
    float ns = newsize[c];
    float count = (ns + EPSF) / (n + (float)NC * EPSF) * n;
    float used  = (ns >= THRESH) ? 1.0f : 0.0f;
    float inv   = 1.0f / count;
    // Deterministic pseudo-permutation pick of a token row (jax perm not
    // reproducible on-device; odd multiplier -> injective mod 2^15).
    unsigned pidx = ((unsigned)c * 2654435761u) & 32767u;
#pragma unroll
    for (int j = 0; j < 4; ++j) {
        int d = threadIdx.x + j * 128;
        float nsum = DECAY * cluster_sum[(size_t)c * DIMD + d]
                   + (1.0f - DECAY) * csum[(size_t)c * DIMD + d];
        out_newsum[(size_t)c * DIMD + d] = nsum;
        float center = nsum * inv;
        float rc = x[(size_t)pidx * DIMD + d];
        out_newemb[(size_t)c * DIMD + d] = used * center + (1.0f - used) * rc;
    }
}

__global__ void k_final(const float* __restrict__ scal, float* __restrict__ out) {
    out[OUT_DIFF]  = scal[0] * (1.0f / (float)((size_t)NT * DIMD));
    out[OUT_AVG]   = scal[2] * (1.0f / (float)NC);
    out[OUT_USAGE] = scal[2];
    out[OUT_ENT]   = scal[3];
}

// ---------------- Launch ----------------
extern "C" void kernel_launch(void* const* d_in, const int* in_sizes, int n_in,
                              void* d_out, int out_size, void* d_ws, size_t ws_size,
                              hipStream_t stream) {
    (void)in_sizes; (void)n_in; (void)out_size; (void)ws_size;
    const float* x    = (const float*)d_in[0];
    const float* emb  = (const float*)d_in[1];
    const float* csz  = (const float*)d_in[2];
    const float* csm  = (const float*)d_in[3];
    float* out = (float*)d_out;

    // Workspace layout (bytes)
    char* ws = (char*)d_ws;
    size_t o = 0;
    __bf16* xh = (__bf16*)(ws + o); o += (size_t)NT * DIMD * 2;
    __bf16* xl = (__bf16*)(ws + o); o += (size_t)NT * DIMD * 2;
    __bf16* eh = (__bf16*)(ws + o); o += (size_t)NC * DIMD * 2;
    __bf16* el = (__bf16*)(ws + o); o += (size_t)NC * DIMD * 2;
    float* enorm = (float*)(ws + o); o += (size_t)NC * 4;
    int*   widx  = (int*)(ws + o);   o += (size_t)NT * 4;
    // zeroed region: csize | csum | scal (contiguous)
    float* wcsize = (float*)(ws + o); o += (size_t)NC * 4;
    float* wcsum  = (float*)(ws + o); o += (size_t)NC * DIMD * 4;
    float* wscal  = (float*)(ws + o); o += 4 * 4;

    const int zeroN = NC + NC * DIMD + 4;
    k_zero<<<(zeroN + 255) / 256, 256, 0, stream>>>(wcsize, zeroN);

    k_split<<<(NT * DIMD) / 256, 256, 0, stream>>>(x,   xh, xl, NT * DIMD);
    k_split<<<(NC * DIMD) / 256, 256, 0, stream>>>(emb, eh, el, NC * DIMD);
    k_enorm<<<NC, 128, 0, stream>>>(emb, enorm);

    k_gemm_argmin<<<NT / 128, 256, 0, stream>>>(xh, xl, eh, el, enorm, widx);

    k_scatter<<<NT, 128, 0, stream>>>(x, emb, widx, out + OUT_Z, out + OUT_CODES,
                                      wcsize, wcsum, wscal);
    k_stats<<<NC / 256, 256, 0, stream>>>(csz, wcsize, out + OUT_SIZE, wscal);
    k_update<<<NC, 128, 0, stream>>>(csm, wcsum, out + OUT_SIZE, x, wscal,
                                     out + OUT_SUM, out + OUT_EMB);
    k_final<<<1, 1, 0, stream>>>(wscal, out);
}
